// CVNet_Rerank_5669356835106
// MI455X (gfx1250) — matrix operand order repaired and checked by reading, not compiled, not run on hardware
//
#include <hip/hip_runtime.h>
#include <hip/hip_bf16.h>

// ---------------------------------------------------------------------------
// CVNet rerank head on gfx1250 (MI455X).  All GEMM-shaped work goes through
// v_wmma_f32_16x16x32_f16 (wave32, one wave per 16x16 C-tile group).
// Conv4d activations are kept channel-last f16 (K-padded) so every WMMA B
// fragment is a single contiguous 32B global_load_b128.
// Workspace budget ~188MB (bump-allocated from d_ws).
// ---------------------------------------------------------------------------

typedef _Float16 h16;
typedef __attribute__((ext_vector_type(16))) _Float16 v16h;
typedef __attribute__((ext_vector_type(8)))  _Float16 v8h;
typedef __attribute__((ext_vector_type(8)))  float    v8f;

#define DEV static __device__ __forceinline__

DEV int imin(int a, int b) { return a < b ? a : b; }

DEV v8f wmma16(v16h a, v16h b, v8f c) {
  // D = A(16x32 f16) x B(32x16 f16) + C(16x16 f32)
  return __builtin_amdgcn_wmma_f32_16x16x32_f16(false, a, false, b, (short)0, c,
                                                false, false);
}

// A fragment: row-major [M][K] f16.  `row` already points at this lane's M row
// (M = lane&15).  khalf = lane>>4.  Lane holds K {kb+kh*16..+15}.
DEV v16h frag_a(const h16* __restrict__ row, int kb, int khalf) {
  v8h lo = *(const v8h*)(row + kb + khalf * 16);
  v8h hi = *(const v8h*)(row + kb + khalf * 16 + 8);
  v16h r;
#pragma unroll
  for (int i = 0; i < 8; ++i) { r[i] = lo[i]; r[8 + i] = hi[i]; }
  return r;
}

// B fragment from K-contiguous B^T [N][K]: lane holds col N = lane&15,
// K range kb + (lane>>4)*16 .. +15  (one 32B contiguous load).
DEV v16h frag_b(const h16* __restrict__ row, int kb, int khalf) {
  return *(const v16h*)(row + kb + khalf * 16);
}

DEV v16h zero16h() {
  v16h r;
#pragma unroll
  for (int i = 0; i < 16; ++i) r[i] = (h16)0.f;
  return r;
}

// ---------------------------------------------------------------------------
// elementwise / prep kernels
// ---------------------------------------------------------------------------

__global__ void k_cast_f32_f16(const float* __restrict__ x, h16* __restrict__ y,
                               long n) {
  long i = (long)blockIdx.x * blockDim.x + threadIdx.x;
  if (i < n) y[i] = (h16)x[i];
}

__global__ void k_fill0_h16(h16* __restrict__ p, long n) {
  long i = (long)blockIdx.x * blockDim.x + threadIdx.x;
  if (i < n) p[i] = (h16)0.f;
}

// bilinear align-corners resize 20x20 -> side x side, [16][256] images
__global__ void k_resize(const float* __restrict__ in, float* __restrict__ out,
                         int side) {
  long idx = (long)blockIdx.x * blockDim.x + threadIdx.x;
  long total = (long)16 * 256 * side * side;
  if (idx >= total) return;
  int x = (int)(idx % side);
  long t = idx / side;
  int y = (int)(t % side);
  long bc = t / side;
  float f = 19.f / (float)(side - 1);
  float py = y * f, px = x * f;
  int y0 = (int)floorf(py); int y1 = imin(y0 + 1, 19); float wy = py - (float)y0;
  int x0 = (int)floorf(px); int x1 = imin(x0 + 1, 19); float wx = px - (float)x0;
  const float* p = in + bc * 400;
  float v00 = p[y0 * 20 + x0], v01 = p[y0 * 20 + x1];
  float v10 = p[y1 * 20 + x0], v11 = p[y1 * 20 + x1];
  float v0 = v00 * (1.f - wx) + v01 * wx;
  float v1 = v10 * (1.f - wx) + v11 * wx;
  out[idx] = v0 * (1.f - wy) + v1 * wy;
}

// im2col for 3x3 pad-1 conv: col[b][p][kk] (f16), kk = ci*9 + kh*3 + kw.
// Rows p in [s*s, Np) zero-filled (Np multiple of 16).
__global__ void k_im2col(const float* __restrict__ x, h16* __restrict__ col,
                         int s, int Np) {
  long idx = (long)blockIdx.x * blockDim.x + threadIdx.x;
  long total = (long)16 * Np * 2304;
  if (idx >= total) return;
  int kk = (int)(idx % 2304);
  long t = idx / 2304;
  int p = (int)(t % Np);
  int b = (int)(t / Np);
  float v = 0.f;
  int s2 = s * s;
  if (p < s2) {
    int ci = kk / 9;
    int r9 = kk - ci * 9;
    int kh = r9 / 3, kw = r9 - kh * 3;
    int py = p / s;
    int y = py + kh - 1;
    int xx = (p - py * s) + kw - 1;
    if ((unsigned)y < (unsigned)s && (unsigned)xx < (unsigned)s)
      v = x[(((size_t)b * 256 + ci) * s + y) * s + xx];
  }
  col[idx] = (h16)v;
}

// per-pixel channel L2 norm of [16][256][s2] -> nrm[b*400+p]
__global__ void k_norms(const float* __restrict__ x, float* __restrict__ nrm,
                        int s2) {
  int idx = blockIdx.x * blockDim.x + threadIdx.x;
  if (idx >= 16 * s2) return;
  int p = idx % s2, b = idx / s2;
  float s = 0.f;
  for (int c = 0; c < 256; ++c) {
    float v = x[((size_t)b * 256 + c) * s2 + p];
    s += v * v;
  }
  nrm[b * 400 + p] = sqrtf(s);
}

// channel-last f16 repack: [16][256][s2] f32 -> [16][Pp][256] f16 (zero pad)
__global__ void k_chanlast(const float* __restrict__ x, h16* __restrict__ chl,
                           int s2, int Pp) {
  long idx = (long)blockIdx.x * blockDim.x + threadIdx.x;
  long total = (long)16 * Pp * 256;
  if (idx >= total) return;
  int c = (int)(idx % 256);
  long t = idx / 256;
  int p = (int)(t % Pp);
  int b = (int)(t / Pp);
  float v = (p < s2) ? x[((size_t)b * 256 + c) * s2 + p] : 0.f;
  chl[idx] = (h16)v;
}

// conv4d weight repack: w[co][ci][k^4] f32 -> Wt[off][co][cip] f16 (ci zero-pad)
__global__ void k_prep_w4d(const float* __restrict__ w, h16* __restrict__ Wt,
                           int Co, int Ci, int k4, int cip) {
  long idx = (long)blockIdx.x * blockDim.x + threadIdx.x;
  long total = (long)k4 * Co * cip;
  if (idx >= total) return;
  int ci = (int)(idx % cip);
  long t = idx / cip;
  int co = (int)(t % Co);
  int off = (int)(t / Co);
  float v = (ci < Ci) ? w[((size_t)co * Ci + ci) * k4 + off] : 0.f;
  Wt[idx] = (h16)v;
}

// ---------------------------------------------------------------------------
// WMMA GEMM kernels
// ---------------------------------------------------------------------------

// C[b][co][n<s2] = sum_k A[co][k] * col[b][n][k].  One wave: 16 N cols x MT*16 co.
template <int MT>
__global__ void k_gemm_conv2d(const h16* __restrict__ A,
                              const h16* __restrict__ Bt,
                              float* __restrict__ out, int K, int Np, int s2,
                              int Co) {
  int lane = threadIdx.x;
  int nt = blockIdx.x, b = blockIdx.y, mg = blockIdx.z;
  int ncol = lane & 15, khalf = lane >> 4;
  const h16* rowb = Bt + ((size_t)b * Np + nt * 16 + ncol) * K;
  v8f acc[MT];
#pragma unroll
  for (int m = 0; m < MT; ++m)
#pragma unroll
    for (int i = 0; i < 8; ++i) acc[m][i] = 0.f;
  for (int kb = 0; kb < K; kb += 32) {
    v16h bf = frag_b(rowb, kb, khalf);
#pragma unroll
    for (int m = 0; m < MT; ++m) {
      const h16* rowa = A + (size_t)(mg * (MT * 16) + m * 16 + ncol) * K;
      acc[m] = wmma16(frag_a(rowa, kb, khalf), bf, acc[m]);
    }
  }
  int n = nt * 16 + ncol;
  if (n < s2) {
#pragma unroll
    for (int m = 0; m < MT; ++m)
#pragma unroll
      for (int r = 0; r < 8; ++r) {
        int co = mg * (MT * 16) + m * 16 + ((lane < 16) ? r : r + 8);
        out[((size_t)b * Co + co) * s2 + n] = acc[m][r];
      }
  }
}

// craw[b][p][q] = sum_c qchl[b][p][c] * kchl[b][q][c]   (K = 256)
__global__ void k_gemm_corr(const h16* __restrict__ Achl,
                            const h16* __restrict__ Bchl,
                            float* __restrict__ craw, int Pp, int Pq, int hs2,
                            int ht2) {
  const int K = 256;
  int lane = threadIdx.x;
  int nt = blockIdx.x, mt = blockIdx.y, b = blockIdx.z;
  int col = lane & 15, khalf = lane >> 4;
  const h16* rowa = Achl + ((size_t)b * Pp + mt * 16 + col) * K;
  const h16* rowb = Bchl + ((size_t)b * Pq + nt * 16 + col) * K;
  v8f acc;
#pragma unroll
  for (int i = 0; i < 8; ++i) acc[i] = 0.f;
  for (int kb = 0; kb < K; kb += 32)
    acc = wmma16(frag_a(rowa, kb, khalf), frag_b(rowb, kb, khalf), acc);
  int q = nt * 16 + col;
  if (q < ht2) {
#pragma unroll
    for (int r = 0; r < 8; ++r) {
      int p = mt * 16 + ((lane < 16) ? r : r + 8);
      if (p < hs2) craw[((size_t)b * hs2 + p) * ht2 + q] = acc[r];
    }
  }
}

// ---------------------------------------------------------------------------
// 4D separable interp (align-corners) + cosine normalize + ReLU.
// Output: corr channel-last [b][pos][16] f16 (channel = pair, 9 valid).
// ---------------------------------------------------------------------------
DEV void interp_iw(int i, int m, int& i0, int& i1, float& w) {
  float pf = (float)i * ((float)(m - 1) / 19.f);
  i0 = (int)floorf(pf);
  i1 = imin(i0 + 1, m - 1);
  w = pf - (float)i0;
}

__global__ void k_interp4d(const float* __restrict__ craw,
                           const float* __restrict__ sn,
                           const float* __restrict__ tn,
                           h16* __restrict__ corr, int hs, int ht, int pair) {
  long idx = (long)blockIdx.x * blockDim.x + threadIdx.x;
  if (idx >= (long)16 * 160000) return;
  int pos = (int)(idx % 160000);
  int b = (int)(idx / 160000);
  int kx = pos % 20; int t = pos / 20;
  int ky = t % 20; t /= 20;
  int qx = t % 20; int qy = t / 20;
  int y0, y1, x0, x1, u0, u1, v0, v1;
  float wy, wx, wu, wv;
  interp_iw(qy, hs, y0, y1, wy);
  interp_iw(qx, hs, x0, x1, wx);
  interp_iw(ky, ht, u0, u1, wu);
  interp_iw(kx, ht, v0, v1, wv);
  int ys[2] = {y0, y1}; float wys[2] = {1.f - wy, wy};
  int xs[2] = {x0, x1}; float wxs[2] = {1.f - wx, wx};
  int us[2] = {u0, u1}; float wus[2] = {1.f - wu, wu};
  int vs[2] = {v0, v1}; float wvs[2] = {1.f - wv, wv};
  int hs2 = hs * hs, ht2 = ht * ht;
  const float* cb = craw + (size_t)b * hs2 * ht2;
  const float* snb = sn + b * 400;
  const float* tnb = tn + b * 400;
  float acc = 0.f;
#pragma unroll
  for (int a = 0; a < 2; ++a)
#pragma unroll
    for (int b2 = 0; b2 < 2; ++b2) {
      float wp = wys[a] * wxs[b2];
      int p = ys[a] * hs + xs[b2];
      float snp = snb[p];
      const float* rp = cb + (size_t)p * ht2;
#pragma unroll
      for (int c2 = 0; c2 < 2; ++c2)
#pragma unroll
        for (int d2 = 0; d2 < 2; ++d2) {
          int q = us[c2] * ht + vs[d2];
          acc += wp * wus[c2] * wvs[d2] * (rp[q] / (snp * tnb[q] + 1e-8f));
        }
    }
  corr[((size_t)b * 160000 + pos) * 16 + pair] = (h16)fmaxf(acc, 0.f);
}

// ---------------------------------------------------------------------------
// Conv4d as offset-loop implicit GEMM over channel-last f16 activations.
// One wave: 16 output positions (N) x MT*16 output channels.  Per kernel
// offset the B fragment is ONE contiguous 32B load (input row stride = cis
// f16 channels, zero-padded).  f32 accumulate, +bias, channel-last f16 out.
// ---------------------------------------------------------------------------
template <int MT, int KCH>
__global__ void k_conv4d(const h16* __restrict__ in, const h16* __restrict__ Wt,
                         const float* __restrict__ bias, h16* __restrict__ out,
                         int I, int O, int kk, int st, int pd, int cis,
                         int cop) {
  const int lane = threadIdx.x;
  const int Co = MT * 16;
  const int cip = KCH * 32;
  int O2 = O * O, O3 = O2 * O, O4 = O3 * O;
  int I4 = I * I * I * I;
  int nt = blockIdx.x;
  int n = nt * 16 + (lane & 15);
  int b = n / O4;
  int r = n - b * O4;
  int ow = r / O3; r -= ow * O3;
  int ox = r / O2; r -= ox * O2;
  int oy = r / O;
  int oz = r - oy * O;
  int bw = ow * st - pd, bx = ox * st - pd, by = oy * st - pd, bz = oz * st - pd;
  const h16* inb = in + (size_t)b * I4 * cis;
  int khalf = lane >> 4;
  v8f acc[MT];
#pragma unroll
  for (int m = 0; m < MT; ++m)
#pragma unroll
    for (int i = 0; i < 8; ++i) acc[m][i] = 0.f;
  int k4 = kk * kk * kk * kk;
  int f0 = 0, f1 = 0, f2 = 0, f3 = 0;
  for (int off = 0; off < k4; ++off) {
    int iw = bw + f0, ix = bx + f1, iy = by + f2, iz = bz + f3;
    bool inbnd = ((unsigned)iw < (unsigned)I) && ((unsigned)ix < (unsigned)I) &&
                 ((unsigned)iy < (unsigned)I) && ((unsigned)iz < (unsigned)I);
    const h16* row =
        inb + (size_t)(((iw * I + ix) * I + iy) * I + iz) * cis;
    const h16* wb = Wt + (size_t)off * Co * cip;
#pragma unroll
    for (int kc = 0; kc < KCH; ++kc) {
      int cb = kc * 32 + khalf * 16;
      v16h bf = (inbnd && cb < cis) ? *(const v16h*)(row + cb) : zero16h();
#pragma unroll
      for (int m = 0; m < MT; ++m) {
        const h16* rowa = wb + (size_t)(m * 16 + (lane & 15)) * cip;
        acc[m] = wmma16(frag_a(rowa, kc * 32, khalf), bf, acc[m]);
      }
    }
    if (++f3 == kk) { f3 = 0; if (++f2 == kk) { f2 = 0; if (++f1 == kk) { f1 = 0; ++f0; } } }
  }
  int pos = ((ow * O + ox) * O + oy) * O + oz;
  size_t rowo = ((size_t)b * O4 + pos) * cop;
#pragma unroll
  for (int m = 0; m < MT; ++m)
#pragma unroll
    for (int rr = 0; rr < 8; ++rr) {
      int co = m * 16 + ((lane < 16) ? rr : rr + 8);
      out[rowo + co] = (h16)(acc[m][rr] + bias[co]);
    }
  if (lane < 16)
    for (int c = Co; c < cop; ++c) out[rowo + c] = (h16)0.f;  // K-pad zeros
}

// GroupNorm(4 groups, f32 stats) + ReLU over channel-last f16 [b][V][cop].
// Grid (16 batches, 4 groups), 256 threads.
__global__ void k_gn_relu_cl(h16* __restrict__ x, const float* __restrict__ gw,
                             const float* __restrict__ gb, int Co, int cop,
                             int V) {
  int b = blockIdx.x, g = blockIdx.y;
  int cg = Co / 4;
  int n = cg * V;
  int tid = threadIdx.x;
  size_t base = (size_t)b * V * cop + (size_t)g * cg;
  float s = 0.f, ss = 0.f;
  for (int i = tid; i < n; i += 256) {
    int pos = i / cg;
    int c = i - pos * cg;
    float v = (float)x[base + (size_t)pos * cop + c];
    s += v;
    ss += v * v;
  }
  __shared__ float sh1[256], sh2[256];
  sh1[tid] = s; sh2[tid] = ss;
  __syncthreads();
  for (int o = 128; o > 0; o >>= 1) {
    if (tid < o) { sh1[tid] += sh1[tid + o]; sh2[tid] += sh2[tid + o]; }
    __syncthreads();
  }
  float mean = sh1[0] / (float)n;
  float var = sh2[0] / (float)n - mean * mean;
  float inv = rsqrtf(var + 1e-5f);
  for (int i = tid; i < n; i += 256) {
    int pos = i / cg;
    int c = i - pos * cg;
    size_t idx = base + (size_t)pos * cop + c;
    float v = ((float)x[idx] - mean) * inv * gw[g * cg + c] + gb[g * cg + c];
    x[idx] = (h16)fmaxf(v, 0.f);
  }
}

// mean-pool + 2-layer MLP + softmax[:,1].  act is channel-last [b][V][128].
__global__ void k_head(const h16* __restrict__ act,
                       const float* __restrict__ w1, const float* __restrict__ b1,
                       const float* __restrict__ w2, const float* __restrict__ b2,
                       float* __restrict__ outp, int V) {
  int b = blockIdx.x, tid = threadIdx.x;
  __shared__ float pooled[128], hbuf[128];
  float s = 0.f;
  for (int i = 0; i < V; ++i)
    s += (float)act[((size_t)b * V + i) * 128 + tid];
  pooled[tid] = s / (float)V;
  __syncthreads();
  float h = b1[tid];
  for (int c = 0; c < 128; ++c) h += w1[tid * 128 + c] * pooled[c];
  hbuf[tid] = fmaxf(h, 0.f);
  __syncthreads();
  if (tid == 0) {
    float l0 = b2[0], l1 = b2[1];
    for (int c = 0; c < 128; ++c) { l0 += w2[c] * hbuf[c]; l1 += w2[128 + c] * hbuf[c]; }
    float mx = fmaxf(l0, l1);
    float e0 = expf(l0 - mx), e1 = expf(l1 - mx);
    outp[b] = e1 / (e0 + e1);
  }
}

// ---------------------------------------------------------------------------
// Host orchestration
// ---------------------------------------------------------------------------
extern "C" void kernel_launch(void* const* d_in, const int* in_sizes, int n_in,
                              void* d_out, int out_size, void* d_ws,
                              size_t ws_size, hipStream_t stream) {
  (void)in_sizes; (void)n_in; (void)out_size; (void)ws_size;
  const float* qft = (const float*)d_in[0];
  const float* kft = (const float*)d_in[1];
  const float* c2w[3]; for (int i = 0; i < 3; ++i) c2w[i] = (const float*)d_in[2 + i];
  const float* c4w[9]; for (int i = 0; i < 9; ++i) c4w[i] = (const float*)d_in[5 + i];
  const float* c4b[9]; for (int i = 0; i < 9; ++i) c4b[i] = (const float*)d_in[14 + i];
  const float* gnw[9]; for (int i = 0; i < 9; ++i) gnw[i] = (const float*)d_in[23 + i];
  const float* gnb[9]; for (int i = 0; i < 9; ++i) gnb[i] = (const float*)d_in[32 + i];
  const float* w1 = (const float*)d_in[41];
  const float* b1 = (const float*)d_in[42];
  const float* w2 = (const float*)d_in[43];
  const float* b2 = (const float*)d_in[44];

  // ---- workspace bump allocator (needs ~188 MB) ----
  char* base = (char*)d_ws;
  size_t off = 0;
  auto alloc = [&](size_t bytes) -> char* {
    char* p = base + off;
    off += (bytes + 255) & ~(size_t)255;
    return p;
  };
  const int sides[3] = {10, 14, 20};
  const int s2v[3] = {100, 196, 400};
  const int Pp16[3] = {112, 208, 400};

  float* ws_resize = (float*)alloc((size_t)16 * 256 * 400 * 4);
  h16* ws_w2d = (h16*)alloc((size_t)256 * 2304 * 2);
  h16* ws_col = (h16*)alloc((size_t)16 * 400 * 2304 * 2);
  float* qsc[2][3];
  for (int t = 0; t < 2; ++t)
    for (int sc = 0; sc < 3; ++sc)
      qsc[t][sc] = (float*)alloc((size_t)16 * 256 * s2v[sc] * 4);
  h16* chl[2][3];
  for (int t = 0; t < 2; ++t)
    for (int sc = 0; sc < 3; ++sc)
      chl[t][sc] = (h16*)alloc((size_t)16 * Pp16[sc] * 256 * 2);
  float* nrmb = (float*)alloc((size_t)2 * 3 * 16 * 400 * 4);
  auto nrm = [&](int t, int sc) { return nrmb + ((size_t)(t * 3 + sc)) * 16 * 400; };
  float* ws_craw = (float*)alloc((size_t)16 * 400 * 400 * 4);
  h16* ws_corr = (h16*)alloc((size_t)16 * 160000 * 16 * 2);  // channel-last [b][pos][16]
  h16* ws_w4 = (h16*)alloc((size_t)81 * 128 * 64 * 2);       // >= 625*16*32 too
  h16* bufA = (h16*)alloc((size_t)16 * 10000 * 32 * 2);      // channel-last acts
  h16* bufB = (h16*)alloc((size_t)16 * 10000 * 32 * 2);

  // ---- per-scale resize -> im2col -> WMMA GEMM conv2d -> norms/chanlast ----
  for (int t = 0; t < 2; ++t) {
    const float* src = t ? kft : qft;
    for (int sc = 0; sc < 3; ++sc) {
      int side = sides[sc], s2 = s2v[sc], Pp = Pp16[sc];
      long tot = (long)16 * 256 * s2;
      k_resize<<<(int)((tot + 255) / 256), 256, 0, stream>>>(src, ws_resize, side);
      long tot2 = (long)16 * Pp * 2304;
      k_im2col<<<(int)((tot2 + 255) / 256), 256, 0, stream>>>(ws_resize, ws_col, side, Pp);
      long nw = (long)256 * 2304;
      k_cast_f32_f16<<<(int)((nw + 255) / 256), 256, 0, stream>>>(c2w[sc], ws_w2d, nw);
      dim3 g(Pp / 16, 16, 4);  // (n-tiles, batch, co-groups of 64)
      k_gemm_conv2d<4><<<g, 32, 0, stream>>>(ws_w2d, ws_col, qsc[t][sc], 2304, Pp, s2, 256);
      k_norms<<<(16 * s2 + 255) / 256, 256, 0, stream>>>(qsc[t][sc], nrm(t, sc), s2);
      long tot3 = (long)16 * Pp * 256;
      k_chanlast<<<(int)((tot3 + 255) / 256), 256, 0, stream>>>(qsc[t][sc], chl[t][sc], s2, Pp);
    }
  }

  // ---- 9 correlation pairs: WMMA GEMM -> normalize+interp4d+ReLU -> corr ----
  {
    long n = (long)16 * 160000 * 16;  // zero corr (pads channels 9..15)
    k_fill0_h16<<<(int)((n + 255) / 256), 256, 0, stream>>>(ws_corr, n);
  }
  for (int si = 0; si < 3; ++si)
    for (int ti = 0; ti < 3; ++ti) {
      int hs = sides[si], ht = sides[ti];
      int hs2 = hs * hs, ht2 = ht * ht;
      dim3 g(Pp16[ti] / 16, Pp16[si] / 16, 16);
      k_gemm_corr<<<g, 32, 0, stream>>>(chl[0][si], chl[1][ti], ws_craw,
                                        Pp16[si], Pp16[ti], hs2, ht2);
      long tot = (long)16 * 160000;
      k_interp4d<<<(int)((tot + 255) / 256), 256, 0, stream>>>(
          ws_craw, nrm(0, si), nrm(1, ti), ws_corr, hs, ht, si * 3 + ti);
    }

  // ---- Conv4d stack (offset-loop implicit GEMM, WMMA, channel-last) ----
  // layer 0: 9->16, k=5, st=2, 20^4 -> 10^4, input = corr [b][pos][16]
  {
    int k4 = 625;
    long tot = (long)k4 * 16 * 32;
    k_prep_w4d<<<(int)((tot + 255) / 256), 256, 0, stream>>>(c4w[0], ws_w4, 16, 9, k4, 32);
    k_conv4d<1, 1><<<10000, 32, 0, stream>>>(ws_corr, ws_w4, c4b[0], bufA,
                                             20, 10, 5, 2, 2, /*cis*/16, /*cop*/32);
    k_gn_relu_cl<<<dim3(16, 4), 256, 0, stream>>>(bufA, gnw[0], gnb[0], 16, 32, 10000);
  }
  auto run_layer = [&](int li, const h16* inb, h16* outb, int Ci, int Co,
                       int kk, int st, int I, int O) {
    int k4 = kk * kk * kk * kk;
    int cip = (Ci > 32) ? 64 : 32;   // K padding (== input row stride cis)
    int cop = (Co < 32) ? 32 : Co;   // output row stride (next layer's cis)
    int MT = Co / 16;
    long tot = (long)k4 * Co * cip;
    k_prep_w4d<<<(int)((tot + 255) / 256), 256, 0, stream>>>(c4w[li], ws_w4, Co, Ci, k4, cip);
    int O4 = O * O * O * O;
    int pd = kk / 2;
    if (MT == 1 && cip == 32)
      k_conv4d<1, 1><<<O4, 32, 0, stream>>>(inb, ws_w4, c4b[li], outb, I, O, kk, st, pd, cip, cop);
    else if (MT == 2 && cip == 32)
      k_conv4d<2, 1><<<O4, 32, 0, stream>>>(inb, ws_w4, c4b[li], outb, I, O, kk, st, pd, cip, cop);
    else if (MT == 4 && cip == 32)
      k_conv4d<4, 1><<<O4, 32, 0, stream>>>(inb, ws_w4, c4b[li], outb, I, O, kk, st, pd, cip, cop);
    else if (MT == 4 && cip == 64)
      k_conv4d<4, 2><<<O4, 32, 0, stream>>>(inb, ws_w4, c4b[li], outb, I, O, kk, st, pd, cip, cop);
    else
      k_conv4d<8, 2><<<O4, 32, 0, stream>>>(inb, ws_w4, c4b[li], outb, I, O, kk, st, pd, cip, cop);
    k_gn_relu_cl<<<dim3(16, 4), 256, 0, stream>>>(outb, gnw[li], gnb[li], Co, cop, O4);
  };
  run_layer(1, bufA, bufB, 16, 16, 3, 1, 10, 10);
  run_layer(2, bufB, bufA, 16, 32, 3, 2, 10, 5);
  run_layer(3, bufA, bufB, 32, 32, 3, 1, 5, 5);
  run_layer(4, bufB, bufA, 32, 32, 3, 1, 5, 5);
  run_layer(5, bufA, bufB, 32, 64, 3, 2, 5, 3);
  run_layer(6, bufB, bufA, 64, 64, 3, 1, 3, 3);
  run_layer(7, bufA, bufB, 64, 64, 3, 1, 3, 3);
  run_layer(8, bufB, bufA, 64, 128, 3, 1, 3, 3);

  // ---- pool + MLP + softmax ----
  k_head<<<16, 128, 0, stream>>>(bufA, w1, b1, w2, b2, (float*)d_out, 81);
}
